// MoERaw_84250078478728
// MI455X (gfx1250) — matrix operand order repaired and checked by reading, not compiled, not run on hardware
//
#include <hip/hip_runtime.h>
#include <hip/hip_bf16.h>
#include <hip/hip_fp16.h>

typedef __attribute__((ext_vector_type(16))) _Float16 v16h;
typedef __attribute__((ext_vector_type(8)))  float    v8f;

// ---------------------------------------------------------------------------
// WMMA fragment helpers (wave32, 16x16x32 f16 -> f32)
// A (16x32, 16-bit): lane m=lane&15; half=lane>>4; elems e<8 -> K=8*half+e,
//                    e>=8 -> K=8*half+8+e  (two contiguous 8-half runs)
// B (32x16, 16-bit): lane N=lane&15; half=lane>>4; elems e -> K same as A
// C/D (16x16 f32):   N=lane&15, M = r + 8*(lane>>4)
// ---------------------------------------------------------------------------
__device__ __forceinline__ v8f wmma_f16(v16h a, v16h b, v8f c) {
  return __builtin_amdgcn_wmma_f32_16x16x32_f16(false, a, false, b, (short)0, c,
                                                false, false);
}

__device__ __forceinline__ v16h a_frag_lds(const _Float16* As, int m0, int lda) {
  const int lane = threadIdx.x & 31;
  const _Float16* p = As + (m0 + (lane & 15)) * lda + ((lane >> 4) * 8);
  union { v16h v; uint2 q[4]; } u;
  u.q[0] = *(const uint2*)(p + 0);
  u.q[1] = *(const uint2*)(p + 4);
  u.q[2] = *(const uint2*)(p + 16);
  u.q[3] = *(const uint2*)(p + 20);
  return u.v;
}

// A fragment from 16B-aligned LDS rows (stride 40 halves): two b128 reads
__device__ __forceinline__ v16h a_frag40(const _Float16* As, int m0) {
  const int lane = threadIdx.x & 31;
  const _Float16* p = As + (m0 + (lane & 15)) * 40 + ((lane >> 4) * 8);
  union { v16h v; uint4 q[2]; } u;
  u.q[0] = *(const uint4*)(p);
  u.q[1] = *(const uint4*)(p + 16);
  return u.v;
}

__device__ __forceinline__ v16h b_frag_lds(const _Float16* Ws, int ldb, int n0) {
  const int lane = threadIdx.x & 31;
  const _Float16* p = Ws + ((lane >> 4) * 8) * ldb + n0 + (lane & 15);
  v16h u;
#pragma unroll
  for (int i = 0; i < 16; ++i) u[i] = p[(i < 8 ? i : i + 8) * ldb];
  return u;
}

// 64B async copy global -> LDS (4x b128). IOFFSET advances both LDS dest and
// memory source (ISA 15.18.3). ASYNCcnt += 4 per call.
#define ASYNC_COPY64(ldsdst, voff, sbase)                                      \
  asm volatile(                                                                \
      "global_load_async_to_lds_b128 %0, %1, %2 offset:0\n\t"                  \
      "global_load_async_to_lds_b128 %0, %1, %2 offset:16\n\t"                 \
      "global_load_async_to_lds_b128 %0, %1, %2 offset:32\n\t"                 \
      "global_load_async_to_lds_b128 %0, %1, %2 offset:48"                     \
      :: "v"(ldsdst), "v"(voff), "s"(sbase) : "memory")

// Non-temporal flavor for the once-streamed fc1 weights (keep L2 for A3).
#define ASYNC_COPY64_NT(ldsdst, voff, sbase)                                   \
  asm volatile(                                                                \
      "global_load_async_to_lds_b128 %0, %1, %2 offset:0 th:TH_LOAD_NT\n\t"    \
      "global_load_async_to_lds_b128 %0, %1, %2 offset:16 th:TH_LOAD_NT\n\t"   \
      "global_load_async_to_lds_b128 %0, %1, %2 offset:32 th:TH_LOAD_NT\n\t"   \
      "global_load_async_to_lds_b128 %0, %1, %2 offset:48 th:TH_LOAD_NT"       \
      :: "v"(ldsdst), "v"(voff), "s"(sbase) : "memory")

// ---------------------------------------------------------------------------
// Utility kernels
// ---------------------------------------------------------------------------
__global__ void k_zero_f32(float* __restrict__ p, int n) {
  int i = blockIdx.x * blockDim.x + threadIdx.x;
  if (i < n) p[i] = 0.f;
}

// Pack OIHW fp32 conv weights -> f16 B-matrix [Kp][Co], K = ci*9+dy*3+dx
__global__ void k_pack_w(const float* __restrict__ w, _Float16* __restrict__ bp,
                         int Co, int Kc, int Kp) {
  int i = blockIdx.x * blockDim.x + threadIdx.x;
  if (i >= Kp * Co) return;
  int k = i / Co, n = i - k * Co;
  float v = (k < Kc) ? w[(size_t)n * Kc + k] : 0.f;
  bp[i] = (_Float16)v;
}

// ---------------------------------------------------------------------------
// conv1: 1->16 ch, 3x3 SAME, ReLU, fp32 math -> f16 NCHW
// ---------------------------------------------------------------------------
__global__ void k_conv1(const float* __restrict__ x, const float* __restrict__ w,
                        const float* __restrict__ bias, _Float16* __restrict__ out) {
  int idx = blockIdx.x * blockDim.x + threadIdx.x;  // B*4096 threads exactly
  int b = idx >> 12, p = idx & 4095, h = p >> 6, wc = p & 63;
  float patch[9];
#pragma unroll
  for (int dy = 0; dy < 3; ++dy)
#pragma unroll
    for (int dx = 0; dx < 3; ++dx) {
      int r = h + dy - 1, c = wc + dx - 1;
      float v = 0.f;
      if (r >= 0 && r < 64 && c >= 0 && c < 64)
        v = x[((size_t)b << 12) + (r << 6) + c];
      patch[dy * 3 + dx] = v;
    }
#pragma unroll
  for (int co = 0; co < 16; ++co) {
    float s = bias[co];
#pragma unroll
    for (int t = 0; t < 9; ++t) s += w[co * 9 + t] * patch[t];
    out[(((size_t)b * 16 + co) << 12) + p] = (_Float16)fmaxf(s, 0.f);
  }
}

// ---------------------------------------------------------------------------
// conv2/conv3 as implicit-GEMM WMMA. One workgroup (128 thr / 4 waves) does one
// image row: M = 64 pixels (4 M-subtiles, one per wave) x N = Cout.
// A-fragments gathered (im2col) from an LDS halo tile [Cin][3][66] (zero pad).
// B read from pre-packed f16 [Kp][Cout].
// ---------------------------------------------------------------------------
template <int Cin, int Cout, int Kact, int Kp>
__global__ void __launch_bounds__(128)
k_conv_gemm(const _Float16* __restrict__ Ain, const _Float16* __restrict__ Bp,
            const float* __restrict__ bias, _Float16* __restrict__ Aout) {
  constexpr int NS = Cout / 16;
  __shared__ _Float16 Ah[Cin * 198];  // [ci][dy(3)][col(66)] halo, col c <-> img col c-1
  const int blk = blockIdx.x;
  const int b = blk >> 6, h = blk & 63;
  const int tid = threadIdx.x, lane = tid & 31, wave = tid >> 5;

  for (int i = tid; i < Cin * 198; i += 128) {
    int ci = i / 198, r = i - ci * 198, dy = r / 66, cc = r - dy * 66;
    int row = h + dy - 1, col = cc - 1;
    _Float16 v = (_Float16)0.f;
    if (row >= 0 && row < 64 && col >= 0 && col < 64)
      v = Ain[(((size_t)b * Cin + ci) << 12) + (row << 6) + col];
    Ah[i] = v;
  }
  __syncthreads();

  const int m0 = wave * 16;            // pixel (w) base for this wave
  const int mw = m0 + (lane & 15);     // this lane's A row (pixel w)
  const int kh = (lane >> 4) * 8;
  v8f z = {};
  v8f c[NS];
#pragma unroll
  for (int i = 0; i < NS; ++i) c[i] = z;

  for (int ks = 0; ks < Kp; ks += 32) {
    v16h a;
#pragma unroll
    for (int e = 0; e < 16; ++e) {
      int kk = ks + kh + (e < 8 ? e : e + 8);
      _Float16 v = (_Float16)0.f;
      if (kk < Kact) {
        int ci = kk / 9, rr = kk - ci * 9, dy = rr / 3, dx = rr - dy * 3;
        v = Ah[ci * 198 + dy * 66 + mw + dx];
      }
      a[e] = v;
    }
#pragma unroll
    for (int ns = 0; ns < NS; ++ns) {
      const _Float16* bp = Bp + (size_t)(ks + kh) * Cout + ns * 16 + (lane & 15);
      union { v16h v; } ub;
#pragma unroll
      for (int e = 0; e < 16; ++e) ub.v[e] = bp[(size_t)(e < 8 ? e : e + 8) * Cout];
      c[ns] = wmma_f16(a, ub.v, c[ns]);
    }
  }

  const int h2 = lane >> 4;
#pragma unroll
  for (int ns = 0; ns < NS; ++ns) {
    int co = ns * 16 + (lane & 15);
    float bv = bias[co];
    union { uint4 q; _Float16 hh[8]; } o;
    v8f cc = c[ns];
#pragma unroll
    for (int r = 0; r < 8; ++r) o.hh[r] = (_Float16)fmaxf(cc[r] + bv, 0.f);
    _Float16* dst = Aout + (((size_t)b * Cout + co) << 12) + (h << 6) + m0 + h2 * 8;
    *(uint4*)dst = o.q;  // 8 consecutive w, 16B aligned
  }
}

// ---------------------------------------------------------------------------
// fc1: C[256,1024] += A3h[256,262144](f16) x fc1_w(f32, NT-streamed).
// grid = (8 N-tiles of 128, 64 K-chunks of 4096). 256 thr / 8 waves.
// Double-buffered LDS filled by GLOBAL_LOAD_ASYNC_TO_LDS_B128 (ASYNCcnt).
// Per k-step a cooperative pass converts the fp32 W tile -> f16 LDS once
// (8 v_cvt_pk per thread), so the WMMA loop reads B straight from f16 LDS
// (no WMMA->VALU WAR hazard NOPs, no 8x duplicated converts).
// Split-K via f32 atomics.
// ---------------------------------------------------------------------------
__global__ void __launch_bounds__(256)
k_fc1(const _Float16* __restrict__ A, const float* __restrict__ W,
      float* __restrict__ acc) {
  constexpr int K = 262144, N = 1024, KCHUNK = 4096, STEPS = KCHUNK / 32;
  __shared__ _Float16 As[2][256 * 40];   // async dest, rows 16B-aligned
  __shared__ float    Wsf[2][32 * 132];  // async dest, fp32 tile
  __shared__ _Float16 Wh[2][32 * 68];    // converted f16 tile
  const int tid = threadIdx.x, lane = tid & 31, wave = tid >> 5;
  const int nBase = blockIdx.x * 128;
  const int kBase = blockIdx.y * KCHUNK;

  // Per-thread async plumbing: thread t stages A row t (64B) and W row t>>3,
  // cols (t&7)*16.. (64B). LDS dest = low 32 bits of flat LDS address.
  const int wk = tid >> 3, wc = (tid & 7) * 16;
  const unsigned ldsA[2] = {(unsigned)(size_t)(const void*)&As[0][tid * 40],
                            (unsigned)(size_t)(const void*)&As[1][tid * 40]};
  const unsigned ldsW[2] = {(unsigned)(size_t)(const void*)&Wsf[0][wk * 132 + wc],
                            (unsigned)(size_t)(const void*)&Wsf[1][wk * 132 + wc]};
  const unsigned voffA = (unsigned)tid * (unsigned)(K * 2);
  const unsigned voffW = (unsigned)(wk * N + wc) * 4u;

  v8f z = {};
  v8f c[2][8];
#pragma unroll
  for (int i = 0; i < 2; ++i)
#pragma unroll
    for (int j = 0; j < 8; ++j) c[i][j] = z;

  auto issue = [&](int step, int buf) {
    const size_t k0 = (size_t)kBase + (size_t)step * 32;
    const unsigned long long sA = (unsigned long long)(const void*)A + k0 * 2;
    const unsigned long long sW =
        (unsigned long long)(const void*)W + (k0 * N + (size_t)nBase) * 4;
    ASYNC_COPY64(ldsA[buf], voffA, sA);      // ASYNCcnt += 4
    ASYNC_COPY64_NT(ldsW[buf], voffW, sW);   // ASYNCcnt += 4
  };

  issue(0, 0);
  issue(1, 1);

  for (int i = 0; i < STEPS; ++i) {
    const int buf = i & 1;
    if (i + 1 < STEPS)
      asm volatile("s_wait_asynccnt 8" ::: "memory");   // current buffer landed
    else
      asm volatile("s_wait_asynccnt 0" ::: "memory");
    __syncthreads();  // all waves' async tiles visible

    {  // cooperative fp32 -> f16 conversion of the W tile (once per element)
      const float* src = &Wsf[buf][wk * 132 + wc];
      uint2* dst = (uint2*)&Wh[buf][wk * 68 + wc];
#pragma unroll
      for (int j = 0; j < 4; ++j) {
        float4 f = *(const float4*)(src + 4 * j);
        union { uint2 u; _Float16 hh[4]; } o;
        o.hh[0] = (_Float16)f.x; o.hh[1] = (_Float16)f.y;
        o.hh[2] = (_Float16)f.z; o.hh[3] = (_Float16)f.w;
        dst[j] = o.u;
      }
    }
    __syncthreads();  // converted tile visible

    const int m0 = wave * 32;
    v16h a0 = a_frag40(As[buf], m0);
    v16h a1 = a_frag40(As[buf], m0 + 16);
#pragma unroll
    for (int ns = 0; ns < 8; ++ns) {
      v16h b = b_frag_lds(Wh[buf], 68, ns * 16);
      c[0][ns] = wmma_f16(a0, b, c[0][ns]);
      c[1][ns] = wmma_f16(a1, b, c[1][ns]);
    }
    __syncthreads();  // everyone done reading buf before it is refilled
    if (i + 2 < STEPS) issue(i + 2, buf);
  }

  const int h2 = lane >> 4;
#pragma unroll
  for (int i = 0; i < 2; ++i)
#pragma unroll
    for (int ns = 0; ns < 8; ++ns) {
      int n = nBase + ns * 16 + (lane & 15);
      v8f cc = c[i][ns];
#pragma unroll
      for (int r = 0; r < 8; ++r) {
        int m = wave * 32 + i * 16 + h2 * 8 + r;
        atomicAdd(acc + (size_t)m * N + n, cc[r]);
      }
    }
}

__global__ void k_bias_relu_h(const float* __restrict__ acc,
                              const float* __restrict__ bias,
                              _Float16* __restrict__ out, int n, int N) {
  int i = blockIdx.x * blockDim.x + threadIdx.x;
  if (i < n) out[i] = (_Float16)fmaxf(acc[i] + bias[i % N], 0.f);
}

// ---------------------------------------------------------------------------
// Generic staged-LDS WMMA GEMM: out[m,n] = act(A[m,:K] * W[:,n] + bias[n]).
// 64x64 tile per WG (256 thr / 8 waves; wave: 1 M-sub x 2 N-subs). blockIdx.z
// indexes experts via strides (0 for plain GEMM).
// ---------------------------------------------------------------------------
__global__ void __launch_bounds__(256)
k_gemm_bias_relu(const _Float16* __restrict__ A, const float* __restrict__ W,
                 const float* __restrict__ bias, _Float16* __restrict__ out,
                 int N, int K, size_t zW, size_t zB, size_t zO, int relu) {
  __shared__ _Float16 As[64 * 36];
  __shared__ _Float16 Ws[32 * 68];
  W += (size_t)blockIdx.z * zW;
  bias += (size_t)blockIdx.z * zB;
  out += (size_t)blockIdx.z * zO;
  const int tid = threadIdx.x, lane = tid & 31, wave = tid >> 5;
  const int nBase = blockIdx.x * 64, mBase = blockIdx.y * 64;
  const int msub = wave >> 1, np = (wave & 1) * 2;
  v8f c0 = {}, c1 = {};

  for (int k0 = 0; k0 < K; k0 += 32) {
    {  // stage A: 64 rows x 32 halves
      int m = tid >> 2, off = (tid & 3) * 8;
      uint4 v = *(const uint4*)(A + (size_t)(mBase + m) * K + k0 + off);
      union { uint4 q4; uint2 q2[2]; } t;
      t.q4 = v;
      uint2* d = (uint2*)(As + m * 36 + off);
      d[0] = t.q2[0];
      d[1] = t.q2[1];
    }
    {  // stage W: 32 rows x 64 cols fp32 -> f16
      int k = tid >> 3, cb = (tid & 7) * 8;
      const float4* g = (const float4*)(W + (size_t)(k0 + k) * N + nBase + cb);
      uint2* d = (uint2*)(Ws + k * 68 + cb);
#pragma unroll
      for (int i = 0; i < 2; ++i) {
        float4 f = g[i];
        union { uint2 u; _Float16 hh[4]; } o;
        o.hh[0] = (_Float16)f.x; o.hh[1] = (_Float16)f.y;
        o.hh[2] = (_Float16)f.z; o.hh[3] = (_Float16)f.w;
        d[i] = o.u;
      }
    }
    __syncthreads();
    v16h a = a_frag_lds(As, msub * 16, 36);
    v16h b0 = b_frag_lds(Ws, 68, np * 16);
    v16h b1 = b_frag_lds(Ws, 68, (np + 1) * 16);
    c0 = wmma_f16(a, b0, c0);
    c1 = wmma_f16(a, b1, c1);
    __syncthreads();
  }

  const int h2 = lane >> 4;
#pragma unroll
  for (int i = 0; i < 2; ++i) {
    v8f cc = i ? c1 : c0;
    int n = nBase + (np + i) * 16 + (lane & 15);
    float bv = bias[n];
#pragma unroll
    for (int r = 0; r < 8; ++r) {
      int m = mBase + msub * 16 + h2 * 8 + r;
      float v = cc[r] + bv;
      if (relu) v = fmaxf(v, 0.f);
      out[(size_t)m * N + n] = (_Float16)v;
    }
  }
}

// ---------------------------------------------------------------------------
// gate: softmax(s2 @ gate_w + b), top-2 + renormalize. 1 block, thread = batch.
// ---------------------------------------------------------------------------
__global__ void __launch_bounds__(256)
k_gate(const _Float16* __restrict__ s2, const float* __restrict__ gw,
       const float* __restrict__ gb, float* __restrict__ gp_out,
       int* __restrict__ tpi, float* __restrict__ tpv) {
  const int b = threadIdx.x;
  float acc[8];
#pragma unroll
  for (int e = 0; e < 8; ++e) acc[e] = gb[e];
  for (int l = 0; l < 256; ++l) {
    float sv = (float)s2[b * 256 + l];
#pragma unroll
    for (int e = 0; e < 8; ++e) acc[e] += sv * gw[l * 8 + e];
  }
  float mx = acc[0];
#pragma unroll
  for (int e = 1; e < 8; ++e) mx = fmaxf(mx, acc[e]);
  float sum = 0.f;
#pragma unroll
  for (int e = 0; e < 8; ++e) { acc[e] = __expf(acc[e] - mx); sum += acc[e]; }
  float inv = 1.f / sum;
#pragma unroll
  for (int e = 0; e < 8; ++e) { acc[e] *= inv; gp_out[b * 8 + e] = acc[e]; }
  int i0 = 0; float v0 = acc[0];
#pragma unroll
  for (int e = 1; e < 8; ++e)
    if (acc[e] > v0) { v0 = acc[e]; i0 = e; }
  int i1 = -1; float v1 = -1.f;
#pragma unroll
  for (int e = 0; e < 8; ++e)
    if (e != i0 && acc[e] > v1) { v1 = acc[e]; i1 = e; }
  float s = v0 + v1;
  tpi[b * 2] = i0; tpi[b * 2 + 1] = i1;
  tpv[b * 2] = v0 / s; tpv[b * 2 + 1] = v1 / s;
}

// ---------------------------------------------------------------------------
// expert head (N=18) for the 2 selected experts + weighted combine -> logits.
// ---------------------------------------------------------------------------
__global__ void k_exp2_combine(const _Float16* __restrict__ eh,
                               const float* __restrict__ w2,
                               const float* __restrict__ b2,
                               const int* __restrict__ tpi,
                               const float* __restrict__ tpv,
                               float* __restrict__ logits) {
  int idx = blockIdx.x * blockDim.x + threadIdx.x;
  if (idx >= 256 * 18) return;
  int b = idx / 18, a = idx - b * 18;
  float outv = 0.f;
#pragma unroll
  for (int j = 0; j < 2; ++j) {
    int e = tpi[b * 2 + j];
    const _Float16* ep = eh + ((size_t)e * 256 + b) * 512;
    const float* wp = w2 + (size_t)e * 512 * 18 + a;
    float s = b2[e * 18 + a];
    for (int k = 0; k < 512; ++k) s += (float)ep[k] * wp[(size_t)k * 18];
    outv += tpv[b * 2 + j] * s;
  }
  logits[idx] = outv;
}

// ---------------------------------------------------------------------------
extern "C" void kernel_launch(void* const* d_in, const int* in_sizes, int n_in,
                              void* d_out, int out_size, void* d_ws, size_t ws_size,
                              hipStream_t stream) {
  const float* pixel   = (const float*)d_in[0];
  const float* c1w     = (const float*)d_in[1];
  const float* c1b     = (const float*)d_in[2];
  const float* c2w     = (const float*)d_in[3];
  const float* c2b     = (const float*)d_in[4];
  const float* c3w     = (const float*)d_in[5];
  const float* c3b     = (const float*)d_in[6];
  const float* fc1w    = (const float*)d_in[7];
  const float* fc1b    = (const float*)d_in[8];
  const float* fc2w    = (const float*)d_in[9];
  const float* fc2b    = (const float*)d_in[10];
  const float* gw      = (const float*)d_in[11];
  const float* gb      = (const float*)d_in[12];
  const float* ew1     = (const float*)d_in[13];
  const float* eb1     = (const float*)d_in[14];
  const float* ew2     = (const float*)d_in[15];
  const float* eb2     = (const float*)d_in[16];
  float* out = (float*)d_out;  // [256*18 logits][256*8 gate_probs]

  char* base = (char*)d_ws;
  size_t off = 0;
  auto alloc = [&](size_t bytes) {
    size_t r = off;
    off = (off + bytes + 255) & ~(size_t)255;
    return r;
  };
  _Float16* A1h = (_Float16*)(base + alloc((size_t)256 * 16 * 4096 * 2));
  _Float16* A2h = (_Float16*)(base + alloc((size_t)256 * 32 * 4096 * 2));
  _Float16* A3h = (_Float16*)(base + alloc((size_t)256 * 64 * 4096 * 2));
  _Float16* W2p = (_Float16*)(base + alloc((size_t)160 * 32 * 2));
  _Float16* W3p = (_Float16*)(base + alloc((size_t)288 * 64 * 2));
  float*    acc1 = (float*)(base + alloc((size_t)256 * 1024 * 4));
  _Float16* s1h = (_Float16*)(base + alloc((size_t)256 * 1024 * 2));
  _Float16* s2h = (_Float16*)(base + alloc((size_t)256 * 256 * 2));
  _Float16* ehh = (_Float16*)(base + alloc((size_t)8 * 256 * 512 * 2));
  int*      tpi = (int*)(base + alloc((size_t)256 * 2 * 4));
  float*    tpv = (float*)(base + alloc((size_t)256 * 2 * 4));

  // independent prep
  k_zero_f32<<<(256 * 1024 + 255) / 256, 256, 0, stream>>>(acc1, 256 * 1024);
  k_pack_w<<<(160 * 32 + 255) / 256, 256, 0, stream>>>(c2w, W2p, 32, 144, 160);
  k_pack_w<<<(288 * 64 + 255) / 256, 256, 0, stream>>>(c3w, W3p, 64, 288, 288);

  // conv stack
  k_conv1<<<4096, 256, 0, stream>>>(pixel, c1w, c1b, A1h);
  k_conv_gemm<16, 32, 144, 160><<<256 * 64, 128, 0, stream>>>(A1h, W2p, c2b, A2h);
  k_conv_gemm<32, 64, 288, 288><<<256 * 64, 128, 0, stream>>>(A2h, W3p, c3b, A3h);

  // fc1 (async double-buffered, split-K atomic) + bias/relu
  k_fc1<<<dim3(8, 64), 256, 0, stream>>>(A3h, fc1w, acc1);
  k_bias_relu_h<<<(256 * 1024 + 255) / 256, 256, 0, stream>>>(acc1, fc1b, s1h,
                                                              256 * 1024, 1024);
  // fc2
  k_gemm_bias_relu<<<dim3(4, 4, 1), 256, 0, stream>>>(s1h, fc2w, fc2b, s2h,
                                                      256, 1024, 0, 0, 0, 1);
  // gate (softmax + top2)
  k_gate<<<1, 256, 0, stream>>>(s2h, gw, gb, out + 256 * 18, tpi, tpv);

  // expert layer 1 (dense over all experts)
  k_gemm_bias_relu<<<dim3(8, 4, 8), 256, 0, stream>>>(
      s2h, ew1, eb1, ehh, 512, 256,
      (size_t)256 * 512, (size_t)512, (size_t)256 * 512, 1);

  // expert head (selected 2 only) + weighted combine
  k_exp2_combine<<<(256 * 18 + 127) / 128, 128, 0, stream>>>(ehh, ew2, eb2, tpi,
                                                             tpv, out);
}
